// GraphConvolution_46566035423426
// MI455X (gfx1250) — compile-verified, matrix operand order checked by reading, not run on hardware
//
#include <hip/hip_runtime.h>

#define DIM 192
#define KCHUNKS 6              // 192 / 32
#define TILE_M 16
#define NWAVES 12              // 192 / 16 output-column tiles
#define BLOCK_THREADS (NWAVES * 32)
#define TILES_PER_BLOCK 8      // M-tiles amortizing the in-register B operands
#define WELEMS (DIM * DIM)     // 36864 elements per weight matrix

typedef __attribute__((ext_vector_type(16))) __bf16 v16bf;
typedef __attribute__((ext_vector_type(8)))  __bf16 v8bf;
typedef __attribute__((ext_vector_type(8)))  float  v8f;

__device__ __forceinline__ unsigned short f2bf(float x) {
    unsigned int u = __builtin_bit_cast(unsigned int, x);
    unsigned int r = u + 0x7FFFu + ((u >> 16) & 1u);   // round-to-nearest-even
    return (unsigned short)(r >> 16);
}

// ---------------------------------------------------------------------------
// Kernel 1: CSR row pointers from the (sorted) edge_dst array via lower_bound.
// rowptr[i] = first edge with dst >= i ; rowptr[n] = E falls out naturally.
// ---------------------------------------------------------------------------
__global__ void build_rowptr(const int* __restrict__ edge_dst, int E, int n,
                             int* __restrict__ rowptr) {
    int i = blockIdx.x * blockDim.x + threadIdx.x;
    if (i > n) return;
    int lo = 0, hi = E;
    while (lo < hi) {
        int mid = (lo + hi) >> 1;
        lo = (edge_dst[mid] < i) ? mid + 1 : lo;
        hi = (edge_dst[mid] < i) ? hi : mid;
    }
    rowptr[i] = lo;
}

// ---------------------------------------------------------------------------
// Kernel 2: convert a 192x192 f32 weight matrix to bf16, pre-swizzled into the
// WMMA 16-bit A-operand layout (lane = row-in-tile; K pairs packed in VGPRs,
// split 8/8 across lane halves). Since out = X @ W^T, the B operand equals W
// itself viewed as an A-matrix.
// Packed index: (((j*KCHUNKS + kk)*32 + lane)*16 + i)
// ---------------------------------------------------------------------------
__global__ void pack_weights(const float* __restrict__ W,
                             unsigned short* __restrict__ Wp) {
    int idx = blockIdx.x * blockDim.x + threadIdx.x;
    if (idx >= WELEMS) return;
    int i    = idx & 15;
    int lane = (idx >> 4) & 31;
    int kk   = (idx >> 9) % KCHUNKS;
    int j    = idx / (16 * 32 * KCHUNKS);
    int row  = j * 16 + (lane & 15);
    int hsel = lane >> 4;
    int k    = kk * 32 + ((i < 8) ? (hsel * 8 + i) : (16 + hsel * 8 + (i - 8)));
    Wp[idx] = f2bf(W[row * DIM + k]);
}

// ---------------------------------------------------------------------------
// Kernel 3: fused GEMM + graph aggregation.
// Block = 384 threads (12 waves); wave j owns columns [16j, 16j+16).
// B1/B2 live in VGPRs for the whole block and are reused across
// TILES_PER_BLOCK consecutive 16-node M-tiles.
// ---------------------------------------------------------------------------
__device__ __forceinline__ v16bf load_a_chunk(const unsigned short* base,
                                              int lane, int kk) {
    // lane%16 = row, lane/16 selects the 8/8 K split within the 32-wide chunk
    const unsigned short* p = base + (lane & 15) * DIM + kk * 32 + (lane >> 4) * 8;
    v8bf lo = *(const v8bf*)p;          // K = kbase .. kbase+7
    v8bf hi = *(const v8bf*)(p + 16);   // K = kbase+16 .. kbase+23
    return __builtin_shufflevector(lo, hi, 0, 1, 2, 3, 4, 5, 6, 7,
                                           8, 9, 10, 11, 12, 13, 14, 15);
}

__global__ __launch_bounds__(BLOCK_THREADS)
void gconv_wmma(const float* __restrict__ X,
                const float* __restrict__ b1,
                const float* __restrict__ b2,
                const int* __restrict__ edge_src,
                const int* __restrict__ rowptr,
                const unsigned short* __restrict__ W1p,
                const unsigned short* __restrict__ W2p,
                float* __restrict__ out, int n) {
    __shared__ __align__(16) unsigned short As[2][TILE_M * DIM]; // [0]=X, [1]=aggr

    const int tid  = threadIdx.x;
    const int lane = tid & 31;
    const int wave = tid >> 5;           // output-column tile 0..11

    // ---- B operands: resident in VGPRs for the whole block (L2-hot packed W)
    v16bf B1[KCHUNKS], B2[KCHUNKS];
#pragma unroll
    for (int kk = 0; kk < KCHUNKS; ++kk) {
        const unsigned short* p1 = W1p + (((wave * KCHUNKS + kk) * 32) + lane) * 16;
        const unsigned short* p2 = W2p + (((wave * KCHUNKS + kk) * 32) + lane) * 16;
        v8bf l1 = *(const v8bf*)p1, h1 = *(const v8bf*)(p1 + 8);
        v8bf l2 = *(const v8bf*)p2, h2 = *(const v8bf*)(p2 + 8);
        B1[kk] = __builtin_shufflevector(l1, h1, 0,1,2,3,4,5,6,7,8,9,10,11,12,13,14,15);
        B2[kk] = __builtin_shufflevector(l2, h2, 0,1,2,3,4,5,6,7,8,9,10,11,12,13,14,15);
    }

    const int col   = wave * 16 + (lane & 15);
    const float bias = b1[col] + b2[col];
    const int mh    = (lane >> 4) * 8;
    const int aggF  = tid % DIM;         // aggregation: feature lane
    const int aggG  = tid / DIM;         // aggregation: node group (0/1)

    for (int t = 0; t < TILES_PER_BLOCK; ++t) {
        const int node0 = (blockIdx.x * TILES_PER_BLOCK + t) * TILE_M; // uniform
        if (node0 >= n) break;                                         // uniform

        // ---- Stage X tile into LDS (f32 -> bf16), coalesced across features
#pragma unroll 2
        for (int e = tid; e < TILE_M * DIM; e += BLOCK_THREADS) {
            int r = e / DIM, c = e - r * DIM;
            int node = node0 + r;
            float v = (node < n) ? X[(long)node * DIM + c] : 0.0f;
            As[0][e] = f2bf(v);
        }

        // ---- Aggregation: 2 groups x 192 feature-threads, 8 nodes each
#pragma unroll
        for (int mi = 0; mi < 8; ++mi) {
            int m = aggG * 8 + mi;
            int node = node0 + m;
            float acc = 0.0f;
            if (node < n) {
                int e0 = rowptr[node], e1 = rowptr[node + 1];
                for (int e = e0; e < e1; ++e)
                    acc += X[(long)edge_src[e] * DIM + aggF];
            }
            As[1][m * DIM + aggF] = f2bf(acc);
        }
        __syncthreads();

        // ---- WMMA: acc = bias ; += Xtile x W1^T ; += AggrTile x W2^T
        v8f acc;
#pragma unroll
        for (int v = 0; v < 8; ++v) acc[v] = bias;

#pragma unroll
        for (int kk = 0; kk < KCHUNKS; ++kk) {
            v16bf ax = load_a_chunk(&As[0][0], lane, kk);
            acc = __builtin_amdgcn_wmma_f32_16x16x32_bf16(false, ax, false, B1[kk],
                                                          (short)0, acc, false, false);
            v16bf ag = load_a_chunk(&As[1][0], lane, kk);
            acc = __builtin_amdgcn_wmma_f32_16x16x32_bf16(false, ag, false, B2[kk],
                                                          (short)0, acc, false, false);
        }

        // ---- Store C/D: VGPR v holds M = v + 8*(lane>>4), N = lane&15
#pragma unroll
        for (int v = 0; v < 8; ++v) {
            int node = node0 + mh + v;
            if (node < n) out[(long)node * DIM + col] = acc[v];
        }
        __syncthreads();   // LDS reuse barrier before next tile's staging
    }
}

// ---------------------------------------------------------------------------
// Host launcher. Inputs: shape_features, W1, b1, W2, b2, edge_dst, edge_src
// ws layout: W1p (36864 u16) | W2p (36864 u16) | rowptr ((n+1) i32)
// ---------------------------------------------------------------------------
extern "C" void kernel_launch(void* const* d_in, const int* in_sizes, int n_in,
                              void* d_out, int out_size, void* d_ws, size_t ws_size,
                              hipStream_t stream) {
    const float* X  = (const float*)d_in[0];
    const float* W1 = (const float*)d_in[1];
    const float* b1 = (const float*)d_in[2];
    const float* W2 = (const float*)d_in[3];
    const float* b2 = (const float*)d_in[4];
    const int* edge_dst = (const int*)d_in[5];
    const int* edge_src = (const int*)d_in[6];
    float* out = (float*)d_out;

    const int n = in_sizes[0] / DIM;
    const int E = in_sizes[5];

    unsigned short* W1p = (unsigned short*)d_ws;
    unsigned short* W2p = W1p + WELEMS;
    int* rowptr = (int*)(W2p + WELEMS);

    pack_weights<<<(WELEMS + 255) / 256, 256, 0, stream>>>(W1, W1p);
    pack_weights<<<(WELEMS + 255) / 256, 256, 0, stream>>>(W2, W2p);
    build_rowptr<<<(n + 1 + 255) / 256, 256, 0, stream>>>(edge_dst, E, n, rowptr);

    const int nodes_per_block = TILE_M * TILES_PER_BLOCK;
    const int blocks = (n + nodes_per_block - 1) / nodes_per_block;
    gconv_wmma<<<blocks, BLOCK_THREADS, 0, stream>>>(X, b1, b2, edge_src, rowptr,
                                                     W1p, W2p, out, n);
}
